// BitNetFFN_87471303951007
// MI455X (gfx1250) — 8-bit matrix operand order repaired and checked by reading, compiled, not run on hardware
//
#include <hip/hip_runtime.h>
#include <cstdint>


#define EPS   1e-5f
#define LNEPS 1e-6f

#define T_TOK 8192
#define H_DIM 2048
#define I_DIM 8192
#define W_ELEMS (16777216.0f)

#define LDS_PITCH 80

typedef __attribute__((ext_vector_type(8))) int v8i;
typedef __attribute__((ext_vector_type(4))) int v4i_;
typedef __attribute__((ext_vector_type(4))) unsigned int v4u;


__device__ __forceinline__ v8i wmma_iu8(v8i a, v8i b, v8i c) {
  return __builtin_amdgcn_wmma_i32_16x16x64_iu8(true, a, true, b, c, false, false);
}


__device__ __forceinline__ uint32_t lds_off(const void* p) {
  return (uint32_t)(uintptr_t)p;
}

__device__ __forceinline__ void tdm_load_tile_2d(uint32_t lds_byte,
                                                 const int8_t* gptr,
                                                 uint32_t tile_w,
                                                 uint32_t tile_h,
                                                 uint64_t row_pitch)
{
  uint64_t ga = (uint64_t)(uintptr_t)gptr;
  v4u g0;
  g0[0] = 1u;
  g0[1] = lds_byte;
  g0[2] = (uint32_t)ga;
  g0[3] = ((uint32_t)(ga >> 32) & 0x01ffffffu) | (2u << 30);

  const uint32_t PAD_EN       = 1u << 20;
  const uint32_t PAD_INTERVAL = 3u << 22;
  const uint32_t PAD_AMOUNT   = 3u << 25;
  uint32_t w0 = PAD_EN | PAD_INTERVAL | PAD_AMOUNT;
  uint32_t w1 = (tile_w & 0xffffu) << 16;
  uint32_t w2 = (tile_w >> 16) | ((tile_h & 0xffffu) << 16);
  uint32_t w3 = (tile_h >> 16) | ((tile_w & 0xffffu) << 16);
  uint32_t w4 = (tile_h & 0xffffu);
  uint32_t w5 = (uint32_t)row_pitch;
  uint32_t w6 = (uint32_t)(row_pitch >> 32) & 0xffffu;
  uint32_t w7 = 0u;
  v8i g1;
  g1[0] = (int)w0; g1[1] = (int)w1; g1[2] = (int)w2; g1[3] = (int)w3;
  g1[4] = (int)w4; g1[5] = (int)w5; g1[6] = (int)w6; g1[7] = (int)w7;
  v4i_ gz = {0, 0, 0, 0};
#if defined(__clang_major__) && (__clang_major__ >= 23)
  v8i gz8 = {0, 0, 0, 0, 0, 0, 0, 0};
  __builtin_amdgcn_tensor_load_to_lds(g0, g1, gz, gz, gz8, 0);
#else
  __builtin_amdgcn_tensor_load_to_lds(g0, g1, gz, gz, 0);
#endif
}

__device__ __forceinline__ v8i lds_a_frag(const int8_t* t, int row_local) {
  int lane = threadIdx.x & 31;
  int m    = lane & 15;
  int kh   = (lane >> 4) << 4;
  const int8_t* p = t + (row_local + m) * LDS_PITCH + kh;
  int2 q0 = *(const int2*)(p);
  int2 q1 = *(const int2*)(p + 8);
  int2 q2 = *(const int2*)(p + 32);
  int2 q3 = *(const int2*)(p + 40);
  v8i f;
  f[0] = q0.x; f[1] = q0.y; f[2] = q1.x; f[3] = q1.y;
  f[4] = q2.x; f[5] = q2.y; f[6] = q3.x; f[7] = q3.y;
  return f;
}

__device__ __forceinline__ v8i lds_b_frag(const int8_t* t, int col_local) {
  int lane = threadIdx.x & 31;
  int n    = lane & 15;
  int kq   = (lane >> 4) << 4;
  const int8_t* p = t + (col_local + n) * LDS_PITCH + kq;
  int4 b0 = *(const int4*)(p);
  int4 b1 = *(const int4*)(p + 32);
  v8i f;
  f[0] = b0.x; f[1] = b0.y; f[2] = b0.z; f[3] = b0.w;
  f[4] = b1.x; f[5] = b1.y; f[6] = b1.z; f[7] = b1.w;
  return f;
}

__device__ __forceinline__ signed char q8(float v, float s, float lo, float hi) {
  return (signed char)fminf(fmaxf(rintf(v * s), lo), hi);
}


__global__ __launch_bounds__(256) void abssum_partial_kernel(
    const float* __restrict__ w, float* __restrict__ partials, int slot) {
  size_t n4 = (size_t)(W_ELEMS / 4.0f);
  float s = 0.f;
  for (size_t i = (size_t)blockIdx.x * blockDim.x + threadIdx.x; i < n4;
       i += (size_t)gridDim.x * blockDim.x) {
    float4 v = ((const float4*)w)[i];
    s += fabsf(v.x) + fabsf(v.y) + fabsf(v.z) + fabsf(v.w);
  }
  __shared__ float red[256];
  red[threadIdx.x] = s;
  __syncthreads();
  for (int st = 128; st > 0; st >>= 1) {
    if ((int)threadIdx.x < st) red[threadIdx.x] += red[threadIdx.x + st];
    __syncthreads();
  }
  if (threadIdx.x == 0) partials[(size_t)slot * 1024 + blockIdx.x] = red[0];
}

__global__ __launch_bounds__(256) void finalize_sums_kernel(
    const float* __restrict__ partials, float* __restrict__ sums) {
  const float* p = partials + (size_t)blockIdx.x * 1024;
  float s = p[threadIdx.x] + p[threadIdx.x + 256] + p[threadIdx.x + 512] +
            p[threadIdx.x + 768];
  __shared__ float red[256];
  red[threadIdx.x] = s;
  __syncthreads();
  for (int st = 128; st > 0; st >>= 1) {
    if ((int)threadIdx.x < st) red[threadIdx.x] += red[threadIdx.x + st];
    __syncthreads();
  }
  if (threadIdx.x == 0) sums[blockIdx.x] = red[0];
}


__global__ __launch_bounds__(256) void weight_quant_kernel(
    const float* __restrict__ w, int8_t* __restrict__ wq,
    const float* __restrict__ sums, int slot) {
  float mean  = fmaxf(sums[slot] / W_ELEMS, EPS);
  float scale = 1.0f / mean;
  size_t i = (size_t)blockIdx.x * blockDim.x + threadIdx.x;
  float4 v = ((const float4*)w)[i];
  char4 q;
  q.x = q8(v.x, scale, -1.f, 1.f);
  q.y = q8(v.y, scale, -1.f, 1.f);
  q.z = q8(v.z, scale, -1.f, 1.f);
  q.w = q8(v.w, scale, -1.f, 1.f);
  ((char4*)wq)[i] = q;
}


__global__ __launch_bounds__(256) void act_quant_kernel(
    const float* __restrict__ x, int8_t* __restrict__ xq,
    float* __restrict__ x_invs) {
  int row = blockIdx.x;
  const float4* xr = (const float4*)(x + (size_t)row * H_DIM);
  float4 v0 = xr[threadIdx.x];
  float4 v1 = xr[threadIdx.x + 256];
  float m = fmaxf(fmaxf(fabsf(v0.x), fabsf(v0.y)), fmaxf(fabsf(v0.z), fabsf(v0.w)));
  m = fmaxf(m, fmaxf(fmaxf(fabsf(v1.x), fabsf(v1.y)), fmaxf(fabsf(v1.z), fabsf(v1.w))));
  __shared__ float red[256];
  red[threadIdx.x] = m;
  __syncthreads();
  for (int st = 128; st > 0; st >>= 1) {
    if ((int)threadIdx.x < st)
      red[threadIdx.x] = fmaxf(red[threadIdx.x], red[threadIdx.x + st]);
    __syncthreads();
  }
  float amax  = fmaxf(red[0], EPS);
  float scale = 127.0f / amax;
  char4* qr = (char4*)(xq + (size_t)row * H_DIM);
  char4 q0, q1;
  q0.x = q8(v0.x, scale, -128.f, 127.f); q0.y = q8(v0.y, scale, -128.f, 127.f);
  q0.z = q8(v0.z, scale, -128.f, 127.f); q0.w = q8(v0.w, scale, -128.f, 127.f);
  q1.x = q8(v1.x, scale, -128.f, 127.f); q1.y = q8(v1.y, scale, -128.f, 127.f);
  q1.z = q8(v1.z, scale, -128.f, 127.f); q1.w = q8(v1.w, scale, -128.f, 127.f);
  qr[threadIdx.x]       = q0;
  qr[threadIdx.x + 256] = q1;
  if (threadIdx.x == 0) x_invs[row] = amax / 127.0f;
}


__global__ __launch_bounds__(256) void gemm_gate_up_kernel(
    const int8_t* __restrict__ xq, const float* __restrict__ x_invs,
    const int8_t* __restrict__ wg, const int8_t* __restrict__ wu,
    const float* __restrict__ sums, float* __restrict__ h) {
  __shared__ int8_t smA[2][128 * LDS_PITCH];
  __shared__ int8_t smG[2][64 * LDS_PITCH];
  __shared__ int8_t smU[2][64 * LDS_PITCH];

  float wg_deq = fmaxf(sums[0] / W_ELEMS, EPS);
  float wu_deq = fmaxf(sums[1] / W_ELEMS, EPS);

  int wid = __builtin_amdgcn_readfirstlane((int)(threadIdx.x >> 5));
  bool producer = (wid == 0);
  int wm = wid >> 1;
  int wn = wid & 1;
  int rowblk = blockIdx.x * 128;
  int colblk = blockIdx.y * 64;
  int row0 = rowblk + wm * 32;
  int col0 = colblk + wn * 32;

  v8i zero = {0, 0, 0, 0, 0, 0, 0, 0};
  v8i accg[2][2] = {{zero, zero}, {zero, zero}};
  v8i accu[2][2] = {{zero, zero}, {zero, zero}};

  const int KT = H_DIM / 64;

  if (producer) {
    tdm_load_tile_2d(lds_off(&smA[0][0]), xq + (size_t)rowblk * H_DIM, 64, 128, H_DIM);
    tdm_load_tile_2d(lds_off(&smG[0][0]), wg + (size_t)colblk * H_DIM, 64, 64, H_DIM);
    tdm_load_tile_2d(lds_off(&smU[0][0]), wu + (size_t)colblk * H_DIM, 64, 64, H_DIM);
  }

  for (int kt = 0; kt < KT; ++kt) {
    int buf = kt & 1;
    if (producer) __builtin_amdgcn_s_wait_tensorcnt(0);
    __syncthreads();
    if (producer && (kt + 1) < KT) {
      int nb = buf ^ 1;
      int k  = (kt + 1) * 64;
      tdm_load_tile_2d(lds_off(&smA[nb][0]), xq + (size_t)rowblk * H_DIM + k, 64, 128, H_DIM);
      tdm_load_tile_2d(lds_off(&smG[nb][0]), wg + (size_t)colblk * H_DIM + k, 64, 64, H_DIM);
      tdm_load_tile_2d(lds_off(&smU[nb][0]), wu + (size_t)colblk * H_DIM + k, 64, 64, H_DIM);
    }
    v8i a0  = lds_a_frag(smA[buf], wm * 32);
    v8i a1  = lds_a_frag(smA[buf], wm * 32 + 16);
    v8i bg0 = lds_b_frag(smG[buf], wn * 32);
    v8i bg1 = lds_b_frag(smG[buf], wn * 32 + 16);
    v8i bu0 = lds_b_frag(smU[buf], wn * 32);
    v8i bu1 = lds_b_frag(smU[buf], wn * 32 + 16);
    accg[0][0] = wmma_iu8(a0, bg0, accg[0][0]);
    accg[0][1] = wmma_iu8(a0, bg1, accg[0][1]);
    accg[1][0] = wmma_iu8(a1, bg0, accg[1][0]);
    accg[1][1] = wmma_iu8(a1, bg1, accg[1][1]);
    accu[0][0] = wmma_iu8(a0, bu0, accu[0][0]);
    accu[0][1] = wmma_iu8(a0, bu1, accu[0][1]);
    accu[1][0] = wmma_iu8(a1, bu0, accu[1][0]);
    accu[1][1] = wmma_iu8(a1, bu1, accu[1][1]);
  }

  int lane = threadIdx.x & 31;
  int colL = lane & 15;
  int rowH = (lane >> 4) * 8;
  for (int mi = 0; mi < 2; ++mi)
    for (int ni = 0; ni < 2; ++ni)
      for (int r = 0; r < 8; ++r) {
        int row = row0 + mi * 16 + rowH + r;
        int col = col0 + ni * 16 + colL;
        float xs = x_invs[row];
        float g  = (float)accg[mi][ni][r] * xs * wg_deq;
        float u  = (float)accu[mi][ni][r] * xs * wu_deq;
        g = fmaxf(g, 0.0f);
        h[(size_t)row * I_DIM + col] = u * g * g;
      }
}


__global__ __launch_bounds__(256) void subln_quant_kernel(
    const float* __restrict__ h, const float* __restrict__ gamma,
    int8_t* __restrict__ aq, float* __restrict__ a_invs) {
  int row = blockIdx.x;
  const float4* hr = (const float4*)(h + (size_t)row * I_DIM);
  const float4* gr = (const float4*)gamma;
  float4 hg[8];
  float ss = 0.f, mx = 0.f;
#pragma unroll
  for (int i = 0; i < 8; ++i) {
    int idx = threadIdx.x + i * 256;
    float4 hv = hr[idx];
    float4 gv = gr[idx];
    ss += hv.x * hv.x + hv.y * hv.y + hv.z * hv.z + hv.w * hv.w;
    float4 t = {hv.x * gv.x, hv.y * gv.y, hv.z * gv.z, hv.w * gv.w};
    hg[i] = t;
    mx = fmaxf(mx, fmaxf(fmaxf(fabsf(t.x), fabsf(t.y)),
                         fmaxf(fabsf(t.z), fabsf(t.w))));
  }
  __shared__ float rs[256];
  __shared__ float rm[256];
  rs[threadIdx.x] = ss;
  rm[threadIdx.x] = mx;
  __syncthreads();
  for (int st = 128; st > 0; st >>= 1) {
    if ((int)threadIdx.x < st) {
      rs[threadIdx.x] += rs[threadIdx.x + st];
      rm[threadIdx.x] = fmaxf(rm[threadIdx.x], rm[threadIdx.x + st]);
    }
    __syncthreads();
  }
  float rms   = rsqrtf(rs[0] / (float)I_DIM + LNEPS);
  float amax  = fmaxf(rm[0] * rms, EPS);
  float scale = 127.0f / amax;
  char4* qr = (char4*)(aq + (size_t)row * I_DIM);
#pragma unroll
  for (int i = 0; i < 8; ++i) {
    float4 t = hg[i];
    char4 q;
    q.x = q8(t.x * rms, scale, -128.f, 127.f);
    q.y = q8(t.y * rms, scale, -128.f, 127.f);
    q.z = q8(t.z * rms, scale, -128.f, 127.f);
    q.w = q8(t.w * rms, scale, -128.f, 127.f);
    qr[threadIdx.x + i * 256] = q;
  }
  if (threadIdx.x == 0) a_invs[row] = amax / 127.0f;
}


__global__ __launch_bounds__(256) void gemm_down_kernel(
    const int8_t* __restrict__ aq, const float* __restrict__ a_invs,
    const int8_t* __restrict__ wd, const float* __restrict__ sums,
    float* __restrict__ out) {
  __shared__ int8_t smA[2][64 * LDS_PITCH];
  __shared__ int8_t smB[2][256 * LDS_PITCH];

  float wd_deq = fmaxf(sums[2] / W_ELEMS, EPS);

  int wid = __builtin_amdgcn_readfirstlane((int)(threadIdx.x >> 5));
  bool producer = (wid == 0);
  int wm = wid >> 2;
  int wn = wid & 3;
  int rowblk = blockIdx.x * 64;
  int colblk = blockIdx.y * 256;
  int row0 = rowblk + wm * 32;
  int col0 = colblk + wn * 64;

  v8i zero = {0, 0, 0, 0, 0, 0, 0, 0};
  v8i acc[2][4] = {{zero, zero, zero, zero}, {zero, zero, zero, zero}};

  const int KT = I_DIM / 64;

  if (producer) {
    tdm_load_tile_2d(lds_off(&smA[0][0]), aq + (size_t)rowblk * I_DIM, 64, 64, I_DIM);
    tdm_load_tile_2d(lds_off(&smB[0][0]), wd + (size_t)colblk * I_DIM, 64, 256, I_DIM);
  }

  for (int kt = 0; kt < KT; ++kt) {
    int buf = kt & 1;
    if (producer) __builtin_amdgcn_s_wait_tensorcnt(0);
    __syncthreads();
    if (producer && (kt + 1) < KT) {
      int nb = buf ^ 1;
      int k  = (kt + 1) * 64;
      tdm_load_tile_2d(lds_off(&smA[nb][0]), aq + (size_t)rowblk * I_DIM + k, 64, 64, I_DIM);
      tdm_load_tile_2d(lds_off(&smB[nb][0]), wd + (size_t)colblk * I_DIM + k, 64, 256, I_DIM);
    }
    v8i a0 = lds_a_frag(smA[buf], wm * 32);
    v8i a1 = lds_a_frag(smA[buf], wm * 32 + 16);
#pragma unroll
    for (int ni = 0; ni < 4; ++ni) {
      v8i b = lds_b_frag(smB[buf], wn * 64 + ni * 16);
      acc[0][ni] = wmma_iu8(a0, b, acc[0][ni]);
      acc[1][ni] = wmma_iu8(a1, b, acc[1][ni]);
    }
  }

  int lane = threadIdx.x & 31;
  int colL = lane & 15;
  int rowH = (lane >> 4) * 8;
  for (int mi = 0; mi < 2; ++mi)
    for (int ni = 0; ni < 4; ++ni)
      for (int r = 0; r < 8; ++r) {
        int row = row0 + mi * 16 + rowH + r;
        int col = col0 + ni * 16 + colL;
        out[(size_t)row * H_DIM + col] =
            (float)acc[mi][ni][r] * a_invs[row] * wd_deq;
      }
}


extern "C" void kernel_launch(void* const* d_in, const int* in_sizes, int n_in,
                              void* d_out, int out_size, void* d_ws,
                              size_t ws_size, hipStream_t stream) {
  (void)in_sizes; (void)n_in; (void)out_size; (void)ws_size;
  const float* x     = (const float*)d_in[0];
  const float* w_g   = (const float*)d_in[1];
  const float* w_u   = (const float*)d_in[2];
  const float* w_d   = (const float*)d_in[3];
  const float* gamma = (const float*)d_in[4];
  float* out = (float*)d_out;

  char* p = (char*)d_ws;
  auto alloc = [&](size_t bytes) -> char* {
    char* r = p;
    p += (bytes + 255) & ~(size_t)255;
    return r;
  };
  float*  sums     = (float*)alloc(3 * sizeof(float));
  float*  partials = (float*)alloc(3 * 1024 * sizeof(float));
  int8_t* xq   = (int8_t*)alloc((size_t)T_TOK * H_DIM);
  int8_t* wqg  = (int8_t*)alloc((size_t)I_DIM * H_DIM);
  int8_t* wqu  = (int8_t*)alloc((size_t)I_DIM * H_DIM);
  int8_t* wqd  = (int8_t*)alloc((size_t)H_DIM * I_DIM);
  float*  x_invs = (float*)alloc((size_t)T_TOK * sizeof(float));
  float*  a_invs = (float*)alloc((size_t)T_TOK * sizeof(float));
  float*  hbuf = (float*)alloc((size_t)T_TOK * I_DIM * sizeof(float));
  int8_t* aq   = (int8_t*)alloc((size_t)T_TOK * I_DIM);

  abssum_partial_kernel<<<1024, 256, 0, stream>>>(w_g, partials, 0);
  abssum_partial_kernel<<<1024, 256, 0, stream>>>(w_u, partials, 1);
  abssum_partial_kernel<<<1024, 256, 0, stream>>>(w_d, partials, 2);
  finalize_sums_kernel<<<3, 256, 0, stream>>>(partials, sums);

  weight_quant_kernel<<<16384, 256, 0, stream>>>(w_g, wqg, sums, 0);
  weight_quant_kernel<<<16384, 256, 0, stream>>>(w_u, wqu, sums, 1);
  weight_quant_kernel<<<16384, 256, 0, stream>>>(w_d, wqd, sums, 2);

  act_quant_kernel<<<T_TOK, 256, 0, stream>>>(x, xq, x_invs);

  gemm_gate_up_kernel<<<dim3(T_TOK / 128, I_DIM / 64), 256, 0, stream>>>(
      xq, x_invs, wqg, wqu, sums, hbuf);

  subln_quant_kernel<<<T_TOK, 256, 0, stream>>>(hbuf, gamma, aq, a_invs);

  gemm_down_kernel<<<dim3(T_TOK / 64, H_DIM / 256), 256, 0, stream>>>(
      aq, a_invs, wqd, sums, out);
}
